// NaLPFactBlock_28252294873241
// MI455X (gfx1250) — matrix operand order repaired and checked by reading, not compiled, not run on hardware
//
#include <hip/hip_runtime.h>
#include <hip/hip_bf16.h>

// MI455X / gfx1250, wave32. Matrix path: v_wmma_f32_16x16x32_f16 (f32 accum).
//
// Pipeline:
//  P) pack_frags: f32 weights -> f16 B-matrix fragments (lane=N, 16 contiguous K
//     per lane per 32-K step) for conv_w, gfcn_w[:, :256] (w_i), gfcn_w[:, 256:] (w_j).
//  1) conv_bn_kernel (1 wave / batch element): build picture tile (gather + concat)
//     in LDS as f16, load A-frags (contiguous K per lane half, as B packing), 16 n-tiles x 8 WMMAs,
//     fused conv_b + BN + ReLU epilogue, write x back out in A-fragment order.
//  2) gfcn_kernel (1 wave per (b, g-tile)): 8 WMMAs for a_i + 8 for a_j, then
//     min over arity (min across 8 acc VGPRs + shfl_xor(16) across lane halves,
//     per the 16x16 f32 C/D layout), out = relu(min_i a_i + min_j a_j + bias).
//     Uses min(relu(a_i+a_j+b)) == relu(min_i a_i + min_j a_j + b)  (relu monotone,
//     a_i+a_j separable) to delete the O(B*16*16*512) pairwise stage entirely.

typedef _Float16 v16h __attribute__((ext_vector_type(16)));
typedef _Float16 v8h  __attribute__((ext_vector_type(8)));
typedef float    v8f  __attribute__((ext_vector_type(8)));

#define BATCH     1024
#define ARITY     16
#define EMB       128
#define NFILT     256
#define NGFCN     512
#define LDS_STRIDE 264   // 256 + 8 f16 pad; keeps 16B alignment (264*2 = 528 = 33*16)

__device__ __forceinline__ v8f wmma_f16(v16h a, v16h b, v8f c) {
  return __builtin_amdgcn_wmma_f32_16x16x32_f16(
      /*neg_a=*/false, a, /*neg_b=*/false, b,
      /*c_mod=*/(short)0, c, /*reuse_a=*/false, /*reuse_b=*/false);
}

// ---- Weight pre-swizzle into WMMA B-fragment order -------------------------
// dst[((nt*kTiles + kt)*32 + lane)*16 + e] = (f16) src[n*rowStride + colOff + k]
//   n = nt*16 + (lane & 15),  k = kt*32 + 16*(lane>>4) + e
// (B order used here, contiguous K: lanes 0-15 = N with K 0-15 across VGPR pairs,
//  lanes 16-31 = same N with K 16-31.)
__global__ void pack_frags(const float* __restrict__ src, int rowStride, int colOff,
                           int kTiles, int total, _Float16* __restrict__ dst) {
  int idx = blockIdx.x * blockDim.x + threadIdx.x;
  if (idx >= total) return;
  int e  = idx & 15;
  int l  = (idx >> 4) & 31;
  int t  = idx >> 9;
  int nt = t / kTiles;
  int kt = t - nt * kTiles;
  int n  = nt * 16 + (l & 15);
  int k  = kt * 32 + 16 * (l >> 4) + e;
  dst[idx] = (_Float16)src[n * rowStride + colOff + k];
}

// ---- Kernel 1: gather + concat + GEMM1 + BN + ReLU, emit x as A-fragments --
__global__ void __launch_bounds__(32)
conv_bn_kernel(const int* __restrict__ role_ids,
               const float* __restrict__ value_embeds,
               const float* __restrict__ role_table,
               const _Float16* __restrict__ convPack,
               const float* __restrict__ conv_b,
               const float* __restrict__ bn_gamma,
               const float* __restrict__ bn_beta,
               const float* __restrict__ bn_mean,
               const float* __restrict__ bn_var,
               _Float16* __restrict__ xPack) {
  __shared__ __align__(16) _Float16 lds[ARITY * LDS_STRIDE];
  const int b    = blockIdx.x;
  const int lane = threadIdx.x;
  const int h    = lane >> 4;      // lane half selects K group
  const int r    = lane & 15;

  // Build picture tile in LDS (f16): [arity][0:128]=role_emb, [128:256]=value.
  for (int idx = lane; idx < ARITY * EMB; idx += 32) {
    int a = idx >> 7, k = idx & 127;
    int rid = role_ids[b * ARITY + a];
    lds[a * LDS_STRIDE + k]       = (_Float16)role_table[rid * EMB + k];
    lds[a * LDS_STRIDE + EMB + k] = (_Float16)value_embeds[(b * ARITY + a) * EMB + k];
  }
  __syncthreads();

  // Load all 8 A-fragments (16x32 f16 each) into registers.
  // A layout: lane half h holds K = 16h..16h+15 within each 32-K step (matches B packing).
  v16h ka[8];
#pragma unroll
  for (int kt = 0; kt < 8; ++kt) {
    const _Float16* p = &lds[r * LDS_STRIDE + kt * 32];
    union { v16h v; v8h half2[2]; } u;
    u.half2[0] = *(const v8h*)(p + 16 * h);
    u.half2[1] = *(const v8h*)(p + 16 * h + 8);
    ka[kt] = u.v;
  }
  __syncthreads();

  // 16 n-tiles of 16 filters; 8 WMMAs each (K=256). Fused BN+ReLU epilogue,
  // results written back into LDS (f16) so we can repack into A-frag order.
  for (int nt = 0; nt < 16; ++nt) {
    v8f acc = {};
#pragma unroll
    for (int kt = 0; kt < 8; ++kt) {
      v16h kb = *(const v16h*)(convPack + (((nt * 8 + kt) * 32) + lane) * 16);
      acc = wmma_f16(ka[kt], kb, acc);
    }
    const int f   = nt * 16 + r;              // this lane's filter column
    const float bias  = conv_b[f];
    const float scale = bn_gamma[f] * rsqrtf(bn_var[f] + 1e-5f);
    const float mn    = bn_mean[f];
    const float bt    = bn_beta[f];
#pragma unroll
    for (int rr = 0; rr < 8; ++rr) {          // C layout: VGPR rr -> M = rr + 8h
      float y = (acc[rr] + bias - mn) * scale + bt;
      y = fmaxf(y, 0.0f);
      lds[(rr + 8 * h) * LDS_STRIDE + f] = (_Float16)y;
    }
  }
  __syncthreads();

  // Repack x (16 x 256 f16) into A-fragment order for GEMM2 (same lane pattern).
#pragma unroll
  for (int kt = 0; kt < 8; ++kt) {
    const _Float16* p = &lds[r * LDS_STRIDE + kt * 32];
    union { v16h v; v8h half2[2]; } u;
    u.half2[0] = *(const v8h*)(p + 16 * h);
    u.half2[1] = *(const v8h*)(p + 16 * h + 8);
    *(v16h*)(xPack + (((b * 8 + kt) * 32) + lane) * 16) = u.v;
  }
}

// ---- Kernel 2: GEMM2 (a_i, a_j) + min-over-arity + relu --------------------
__global__ void __launch_bounds__(128)
gfcn_kernel(const _Float16* __restrict__ xPack,
            const _Float16* __restrict__ wiPack,
            const _Float16* __restrict__ wjPack,
            const float* __restrict__ gfcn_b,
            float* __restrict__ out) {
  const int b    = blockIdx.x;
  const int wave = threadIdx.x >> 5;
  const int lane = threadIdx.x & 31;
  const int gt   = blockIdx.y * 4 + wave;   // g-tile 0..31

  v16h ka[8];
#pragma unroll
  for (int kt = 0; kt < 8; ++kt)
    ka[kt] = *(const v16h*)(xPack + (((b * 8 + kt) * 32) + lane) * 16);

  v8f ai = {}, aj = {};
#pragma unroll
  for (int kt = 0; kt < 8; ++kt) {
    v16h bi = *(const v16h*)(wiPack + (((gt * 8 + kt) * 32) + lane) * 16);
    ai = wmma_f16(ka[kt], bi, ai);
    v16h bj = *(const v16h*)(wjPack + (((gt * 8 + kt) * 32) + lane) * 16);
    aj = wmma_f16(ka[kt], bj, aj);
  }

  // Min over arity: per-lane min over 8 acc VGPRs (M = 8h..8h+7), then
  // combine lane halves (lane l <-> l^16 share the same N, other M half).
  float mi = ai[0], mj = aj[0];
#pragma unroll
  for (int rr = 1; rr < 8; ++rr) {
    mi = fminf(mi, ai[rr]);
    mj = fminf(mj, aj[rr]);
  }
  mi = fminf(mi, __shfl_xor(mi, 16, 32));
  mj = fminf(mj, __shfl_xor(mj, 16, 32));

  if (lane < 16) {
    const int g = gt * 16 + lane;
    out[b * NGFCN + g] = fmaxf(mi + mj + gfcn_b[g], 0.0f);
  }
}

extern "C" void kernel_launch(void* const* d_in, const int* in_sizes, int n_in,
                              void* d_out, int out_size, void* d_ws, size_t ws_size,
                              hipStream_t stream) {
  const int*   role_ids     = (const int*)d_in[0];
  const float* value_embeds = (const float*)d_in[1];
  const float* role_table   = (const float*)d_in[2];
  const float* conv_w       = (const float*)d_in[3];
  const float* conv_b       = (const float*)d_in[4];
  const float* bn_gamma     = (const float*)d_in[5];
  const float* bn_beta      = (const float*)d_in[6];
  const float* bn_mean      = (const float*)d_in[7];
  const float* bn_var       = (const float*)d_in[8];
  const float* gfcn_w       = (const float*)d_in[9];
  const float* gfcn_b       = (const float*)d_in[10];
  float* out = (float*)d_out;

  // Workspace layout (f16 elements): total ~9.1 MB.
  _Float16* convPack = (_Float16*)d_ws;                 //  16*8*32*16 =   65536
  _Float16* wiPack   = convPack + 65536;                //  32*8*32*16 =  131072
  _Float16* wjPack   = wiPack + 131072;                 //              131072
  _Float16* xPack    = wjPack + 131072;                 // 1024*8*32*16 = 4194304

  pack_frags<<<65536 / 256, 256, 0, stream>>>(conv_w, 2 * EMB, 0,   8, 65536,  convPack);
  pack_frags<<<131072 / 256, 256, 0, stream>>>(gfcn_w, 2 * NFILT, 0,   8, 131072, wiPack);
  pack_frags<<<131072 / 256, 256, 0, stream>>>(gfcn_w, 2 * NFILT, 256, 8, 131072, wjPack);

  conv_bn_kernel<<<BATCH, 32, 0, stream>>>(role_ids, value_embeds, role_table,
                                           convPack, conv_b, bn_gamma, bn_beta,
                                           bn_mean, bn_var, xPack);

  gfcn_kernel<<<dim3(BATCH, 8), 128, 0, stream>>>(xPack, wiPack, wjPack, gfcn_b, out);
}
